// SingleDimHistLayer_9225589752148
// MI455X (gfx1250) — compile-verified, matrix operand order checked by reading, not run on hardware
//
#include <hip/hip_runtime.h>

// CDNA5 (gfx1250) soft-histogram with WMMA-offloaded reduction.
// out[b,k] = (1/N) * sum_n [ sigmoid(2.5*(t-k)) - sigmoid(2.5*(t-k-1)) ],  t = x*256
//
// Round-3: revert the A-operand double-buffering from round 2 — it raised
// register pressure, broke the compiler's v_pk_mul/v_cvt_pk pairings and grew
// the loop body (1995 -> 2165 instrs) without removing the (TRANS-hazard)
// v_nops. This is the round-1 schedule, which was the tightest.

typedef _Float16 v16h __attribute__((ext_vector_type(16)));
typedef float    v8f  __attribute__((ext_vector_type(8)));

namespace {
constexpr int kBins  = 256;
constexpr int kBatch = 8;
constexpr int kN     = 512 * 512;          // elements per batch
constexpr int kGroups = 8;                 // 16-bin groups per wave (128 bins)
constexpr int kGsets  = 2;                 // 2 group-sets cover 256 bins
constexpr int kTiles  = 128;               // element tiles per (batch, gset)
constexpr int kEpw    = kN / kTiles;       // 2048 elements per wave
constexpr int kWavesPerBlock = 8;          // 256 threads/block (8 wave32)

constexpr float kS   = 3.6067376022224087f;  // 2.5 * log2(e)  (exp2-domain scale)
constexpr float kUsc = kS * 256.0f;          // applied to x:  u = kS * t
constexpr float kC   = 0.0820849986238988f;  // exp(-2.5)
constexpr float kNum = 1.0f - kC;            // 0.91791...
}

// sigma(a) - sigma(a-2.5) with a = 2.5*(t - bin), computed in exp2 domain.
// = ea*(1-c) / ((1+ea)*(1+c*ea)).  Clamp keeps ea finite; overflow of den
// gives rcp(inf)=0 -> g=0 which is the correct limit.
__device__ __forceinline__ float soft_bin(float u, float sb) {
  float arg = u - sb;                              // folds into v_fma with kUsc
  arg = fminf(fmaxf(arg, -112.0f), 112.0f);        // v_med3_num_f32
  float ea  = __builtin_amdgcn_exp2f(arg);         // v_exp_f32
  float den = (1.0f + ea) * fmaf(ea, kC, 1.0f);
  return (ea * kNum) * __builtin_amdgcn_rcpf(den); // v_rcp_f32
}

__global__ __launch_bounds__(kWavesPerBlock * 32)
void softhist_wmma(const float* __restrict__ x, float* __restrict__ out) {
  const int lane = threadIdx.x & 31;
  const int w    = blockIdx.x * kWavesPerBlock + (threadIdx.x >> 5);
  const int tile = w % kTiles;
  const int gset = (w / kTiles) % kGsets;
  const int b    = w / (kTiles * kGsets);

  const int mrow = lane & 15;   // A-matrix row (bin within 16-bin group)
  const int half = lane >> 4;   // 0: needs elements {0..7,16..23}; 1: {8..15,24..31}

  // Per-lane bin constants in exp2 domain: arg = kS*t - kS*bin
  float sbin[kGroups];
#pragma unroll
  for (int g = 0; g < kGroups; ++g)
    sbin[g] = kS * (float)(gset * (kGroups * 16) + g * 16 + mrow);

  v8f acc[kGroups] = {};        // f32 C/D accumulators (all columns identical)

  v16h ones;                    // B = 32x16 all-ones (layout-invariant)
#pragma unroll
  for (int i = 0; i < 16; ++i) ones[i] = (_Float16)1.0f;

  const float* xw = x + (size_t)b * kN + (size_t)tile * kEpw;

#pragma unroll 1
  for (int c0 = 0; c0 < kEpw; c0 += 32) {
    // Each half-wave loads its A-operand element subset (coalesced/broadcast).
    const float4* q = (const float4*)(xw + c0);
    float4 q0 = q[half * 2 + 0];   // elements half*8 + 0..3
    float4 q1 = q[half * 2 + 1];   // elements half*8 + 4..7
    float4 q2 = q[half * 2 + 4];   // elements half*8 + 16..19
    float4 q3 = q[half * 2 + 5];   // elements half*8 + 20..23
    float u[16] = {q0.x, q0.y, q0.z, q0.w, q1.x, q1.y, q1.z, q1.w,
                   q2.x, q2.y, q2.z, q2.w, q3.x, q3.y, q3.z, q3.w};
#pragma unroll
    for (int j = 0; j < 16; ++j) u[j] *= kUsc;

#pragma unroll
    for (int g = 0; g < kGroups; ++g) {
      v16h a;                     // A = 16(bins) x 32(elements) membership tile
#pragma unroll
      for (int j = 0; j < 16; ++j)
        a[j] = (_Float16)soft_bin(u[j], sbin[g]);   // RNE f16, pairs -> cvt_pk

      // D = A x ones + C : per-bin partial sums, f32 accumulate on matrix pipe
      acc[g] = __builtin_amdgcn_wmma_f32_16x16x32_f16(
          /*neg_a=*/false, a, /*neg_b=*/false, ones,
          /*c_mod=*/(short)0, acc[g], /*reuse_a=*/false, /*reuse_b=*/false);
    }
  }

  // C/D layout: lanes 0-15 hold M=0..7 in acc[ ][0..7]; lanes 16-31 hold M=8..15.
  const float invN = 1.0f / (float)kN;
  float* ob = out + b * kBins + gset * (kGroups * 16);
#pragma unroll
  for (int g = 0; g < kGroups; ++g) {
    if (lane == g) {
#pragma unroll
      for (int i = 0; i < 8; ++i)
        atomicAdd(ob + g * 16 + i, acc[g][i] * invN);
    }
    if (lane == 16 + g) {
#pragma unroll
      for (int i = 0; i < 8; ++i)
        atomicAdd(ob + g * 16 + 8 + i, acc[g][i] * invN);
    }
  }
}

extern "C" void kernel_launch(void* const* d_in, const int* in_sizes, int n_in,
                              void* d_out, int out_size, void* d_ws, size_t ws_size,
                              hipStream_t stream) {
  const float* x = (const float*)d_in[0];
  float* out = (float*)d_out;

  // Accumulated via atomics -> must start from zero every call (capture-safe).
  hipMemsetAsync(out, 0, (size_t)out_size * sizeof(float), stream);

  const int waves = kBatch * kGsets * kTiles;          // 2048 waves
  dim3 grid(waves / kWavesPerBlock);                   // 256 blocks
  dim3 block(kWavesPerBlock * 32);                     // 256 threads
  softhist_wmma<<<grid, block, 0, stream>>>(x, out);
}
